// NeuralUniLasso_26061861552588
// MI455X (gfx1250) — matrix operand order repaired and checked by reading, not compiled, not run on hardware
//
#include <hip/hip_runtime.h>

typedef __attribute__((ext_vector_type(16))) _Float16 v16h;
typedef __attribute__((ext_vector_type(8)))  _Float16 v8h;
typedef __attribute__((ext_vector_type(8)))  float    v8f;

#define BROWS 32768
#define NF    256
#define NH1   16
#define NH2   8
#define NPAIR (NF / 2)
#define CHUNK 64                 // features staged per async chunk
#define NBLK  (NF / CHUNK)       // 4
#define XROWS 128                // batch rows per block
#define XSTR  68                 // padded LDS row stride (floats): bank-conflict-free

__device__ __forceinline__ float elu_f(float v) {
    return v > 0.0f ? v : (__expf(v) - 1.0f);
}

// ---------------------------------------------------------------------------
// Transposed formulation: one 16x16x32 f16 WMMA per feature pair computes
//   D[M=h2 channel (f0:0..7 | f1:8..15), N=batch(16)] = A x B + C
//   A[M,K] = W2[f0, K, M]      (M<8, K<16)   | W2[f1, K-16, M-8] (M>=8, K>=16)
//   B[K,N] = h1_f0[N, K] (K<16) | h1_f1[N, K-16] (K>=16)
//   C      = b2 (per channel)
// Lane L then holds all 8 h2 channels of its OWN feature / batch row in the 8
// accumulator VGPRs -> the W3 dot product is a pure in-lane reduction.
// ---------------------------------------------------------------------------
__global__ void pack_w2_kernel(const float* __restrict__ W2,
                               _Float16* __restrict__ W2h) {
    const int p    = blockIdx.x;     // feature pair
    const int L    = threadIdx.x;    // 0..31
    const int n    = L & 15;         // A-matrix M row
    const int half = L >> 4;
    const int f0   = 2 * p;
    const int f1   = f0 + 1;

    v8h lo, hi;
    #pragma unroll
    for (int e = 0; e < 8; ++e) {
        const int fs = half ? f1 : f0;   // lane half selects feature (K = half*16 + e)
        const bool on = half ? (n >= 8) : (n < 8);
        lo[e] = on
              ? (_Float16)W2[(size_t)fs * NH1 * NH2 + e * NH2 + (n & 7)]
              : (_Float16)0.0f;
        hi[e] = on
              ? (_Float16)W2[(size_t)fs * NH1 * NH2 + (e + 8) * NH2 + (n & 7)]
              : (_Float16)0.0f;
    }
    v8h* dst = (v8h*)(W2h + ((size_t)p * 32 + L) * 16);
    dst[0] = lo;
    dst[1] = hi;
}

__global__ void weights_kernel(const float* __restrict__ theta,
                               float* __restrict__ w_dout,
                               float* __restrict__ w_ws) {
    const int f = threadIdx.x;
    const float w = log1pf(__expf(theta[f]));
    w_dout[f] = w;
    w_ws[f]  = w;
}

__launch_bounds__(256)
__global__ void unilasso_main_kernel(
    const float* __restrict__ x,
    const float* __restrict__ W1,
    const float* __restrict__ b1,
    const float* __restrict__ b2,
    const float* __restrict__ W3,
    const float* __restrict__ b3,
    const float* __restrict__ bias,
    const _Float16* __restrict__ W2h,
    const float* __restrict__ wts,
    float* __restrict__ y_out,
    float* __restrict__ Z_out)
{
    // Double-buffered async x staging tiles + per-wave Z staging tile.
    // Row stride 68 floats keeps both b32 column access (bank = (4r+c)%64) and
    // float4 flush access conflict-free; rows stay 16B-aligned (272B).
    __shared__ float XBuf[2][XROWS][XSTR];   // 2 x 34 KiB
    __shared__ float Zbuf[8][16][XSTR];      // 34 KiB

    const int tid  = threadIdx.x;
    const int wave = tid >> 5;
    const int lane = tid & 31;
    const int n    = lane & 15;         // batch row within tile (B-operand N col)
    const int half = lane >> 4;         // 0 -> even feature, 1 -> odd feature
    const int tile = (blockIdx.x * 8 + wave) * 16;

    float (*zb)[XSTR] = Zbuf[wave];
    const float bias0 = bias[0];
    float yacc = 0.0f;

    // ---- async stage of one 128-row x 64-feature x chunk into LDS ----------
    auto stage = [&](int cblk, int bufsel) {
        const size_t gbase = (size_t)blockIdx.x * XROWS * NF + (size_t)cblk * CHUNK;
        #pragma unroll
        for (int i = 0; i < 8; ++i) {
            const int idx = i * 256 + tid;      // 2048 b128 lane-transfers total
            const int row = idx >> 4;           // 16 x 16B groups per row
            const int grp = idx & 15;
            const float* gp = x + gbase + (size_t)row * NF + grp * 4;
            const unsigned ldsa = (unsigned)(uintptr_t)&XBuf[bufsel][row][grp * 4];
            asm volatile("global_load_async_to_lds_b128 %0, %1, off"
                         :: "v"(ldsa), "v"(gp) : "memory");
        }
    };

    stage(0, 0);
    asm volatile("s_wait_asynccnt 0x0" ::: "memory");
    __syncthreads();

    for (int blk = 0; blk < NBLK; ++blk) {
        const int bufsel = blk & 1;
        if (blk + 1 < NBLK) stage(blk + 1, (blk + 1) & 1);   // overlap with compute

        #pragma unroll 2
        for (int q = 0; q < CHUNK / 2; ++q) {
            const int f0 = blk * CHUNK + 2 * q;
            const int f  = f0 + half;            // this lane's feature
            const int col = 2 * q + half;

            const float xv = XBuf[bufsel][wave * 16 + n][col];

            // W1/b1: 16 contiguous f32 each for this lane's feature.
            float w1v[16], b1v[16];
            {
                const float4* wp = (const float4*)(W1 + f * NH1);
                const float4* bp = (const float4*)(b1 + f * NH1);
                *(float4*)&w1v[0]  = wp[0]; *(float4*)&w1v[4]  = wp[1];
                *(float4*)&w1v[8]  = wp[2]; *(float4*)&w1v[12] = wp[3];
                *(float4*)&b1v[0]  = bp[0]; *(float4*)&b1v[4]  = bp[1];
                *(float4*)&b1v[8]  = bp[2]; *(float4*)&b1v[12] = bp[3];
            }

            // B fragment: all 16 h1 values of this lane's feature
            // (element e <-> K = half*16 + e, exactly this feature's K block).
            v16h bfrag;
            #pragma unroll
            for (int e = 0; e < 16; ++e)
                bfrag[e] = (_Float16)elu_f(fmaf(xv, w1v[e], b1v[e]));

            // A fragment: prepacked block-diagonal W2 (32 bytes per lane).
            const int p = f0 >> 1;
            const v8h* ap = (const v8h*)(W2h + ((size_t)p * 32 + lane) * 16);
            const v8h alo = ap[0];
            const v8h ahi = ap[1];
            const v16h afrag = __builtin_shufflevector(
                alo, ahi, 0, 1, 2, 3, 4, 5, 6, 7, 8, 9, 10, 11, 12, 13, 14, 15);

            // C seed: b2 channels of this lane's feature (c[j] = b2[f, j]).
            float b2v[8];
            {
                const float4* bp = (const float4*)(b2 + f * NH2);
                *(float4*)&b2v[0] = bp[0]; *(float4*)&b2v[4] = bp[1];
            }
            v8f c = {b2v[0], b2v[1], b2v[2], b2v[3],
                     b2v[4], b2v[5], b2v[6], b2v[7]};

            c = __builtin_amdgcn_wmma_f32_16x16x32_f16(false, afrag, false, bfrag,
                                                       (short)0, c, false, false);

            // Z = sum_j elu(h2pre[j]) * W3[f, j] + b3[f] -- in-lane reduction.
            float w3v[8];
            {
                const float4* wp = (const float4*)(W3 + f * NH2);
                *(float4*)&w3v[0] = wp[0]; *(float4*)&w3v[4] = wp[1];
            }
            float zv = b3[f];
            #pragma unroll
            for (int j = 0; j < 8; ++j)
                zv = fmaf(elu_f(c[j]), w3v[j], zv);

            yacc = fmaf(zv, wts[f], yacc);
            zb[n][col] = zv;                 // one unconditional b32 store
        }

        // Flush this wave's 16x64 Z tile with coalesced row-runs.
        {
            const int fbase = blk * CHUNK;
            const float4* src = (const float4*)&zb[n][half * 32];
            float4* dst = (float4*)(Z_out + (size_t)(tile + n) * NF + fbase + half * 32);
            #pragma unroll
            for (int qq = 0; qq < 8; ++qq) dst[qq] = src[qq];
        }

        if (blk + 1 < NBLK)
            asm volatile("s_wait_asynccnt 0x0" ::: "memory");
        __syncthreads();
    }

    // y[tile+n] = even-feature partial (lanes 0..15) + odd (lanes 16..31).
    const float yoth = __shfl_xor(yacc, 16, 32);
    if (half == 0) y_out[tile + n] = yacc + yoth + bias0;
}

extern "C" void kernel_launch(void* const* d_in, const int* in_sizes, int n_in,
                              void* d_out, int out_size, void* d_ws, size_t ws_size,
                              hipStream_t stream) {
    (void)in_sizes; (void)n_in; (void)out_size; (void)ws_size;

    const float* x     = (const float*)d_in[0];
    const float* W1    = (const float*)d_in[1];
    const float* b1    = (const float*)d_in[2];
    const float* W2    = (const float*)d_in[3];
    const float* b2    = (const float*)d_in[4];
    const float* W3    = (const float*)d_in[5];
    const float* b3    = (const float*)d_in[6];
    const float* theta = (const float*)d_in[7];
    const float* bias  = (const float*)d_in[8];

    float* out   = (float*)d_out;
    float* y_out = out;                  // [B]
    float* w_out = out + BROWS;          // [F]
    float* Z_out = out + BROWS + NF;     // [B, F]

    _Float16* W2h = (_Float16*)d_ws;                  // 128 pairs * 32 lanes * 32B = 128 KiB
    float*    wts = (float*)((char*)d_ws + 131072);   // [F] softplus(theta)

    pack_w2_kernel<<<NPAIR, 32, 0, stream>>>(W2, W2h);
    weights_kernel<<<1, NF, 0, stream>>>(theta, w_out, wts);
    unilasso_main_kernel<<<BROWS / XROWS, 256, 0, stream>>>(
        x, W1, b1, b2, W3, b3, bias, W2h, wts, y_out, Z_out);
}
